// WTKOContrastiveVAE_11940009083406
// MI455X (gfx1250) — matrix operand order repaired and checked by reading, not compiled, not run on hardware
//
#include <hip/hip_runtime.h>
#include <hip/hip_bf16.h>
#include <math.h>

typedef __attribute__((ext_vector_type(16))) _Float16 v16h;
typedef __attribute__((ext_vector_type(8)))  _Float16 v8h;
typedef __attribute__((ext_vector_type(8)))  float    v8f;

#define NROWS 4096
#define GDIM  2000
#define KPAD  2016   // 63 * 32
#define HDIM  128
#define HCMB  256    // enc(128) + kap(128) fused hidden columns
#define LDIM  10
#define KCLS  10
#define INV_TAU 10.0f

// ---------------------------------------------------------------- reductions
__device__ __forceinline__ float waveReduceSum(float v) {
#pragma unroll
    for (int off = 16; off > 0; off >>= 1) v += __shfl_down(v, off, 32);
    return v;
}

__device__ __forceinline__ float blockReduceSum128(float v) {
    v = waveReduceSum(v);
    __shared__ float sm[4];
    int lane = threadIdx.x & 31, w = threadIdx.x >> 5;
    __syncthreads();
    if (lane == 0) sm[w] = v;
    __syncthreads();
    return sm[0] + sm[1] + sm[2] + sm[3];
}

// ---------------------------------------------------------------- init
__global__ void init_accum(float* scal, int* cw, int* ck, int* ca,
                           float* sums, float* devc) {
    int t = threadIdx.x;
    if (t < 8)   scal[t] = 0.f;
    if (t < 16)  { cw[t] = 0; ck[t] = 0; ca[t] = 0; devc[t] = 0.f; }
    if (t < KCLS * LDIM) sums[t] = 0.f;
}

// ---------------------------------------------------------------- casts
__global__ void cast_pad_x(const float* __restrict__ x, _Float16* __restrict__ o) {
    int idx = blockIdx.x * blockDim.x + threadIdx.x;
    if (idx >= NROWS * KPAD) return;
    int i = idx / KPAD, k = idx % KPAD;
    o[idx] = (k < GDIM) ? (_Float16)x[(size_t)i * GDIM + k] : (_Float16)0.f;
}

// enc_w1, kap_w1: [GDIM x HDIM] -> combined B^T [HCMB x KPAD]
__global__ void transpose_w_pair(const float* __restrict__ we,
                                 const float* __restrict__ wk,
                                 _Float16* __restrict__ o) {
    int idx = blockIdx.x * blockDim.x + threadIdx.x;
    if (idx >= HCMB * KPAD) return;
    int h = idx / KPAD, k = idx % KPAD;
    float v = 0.f;
    if (k < GDIM)
        v = (h < HDIM) ? we[(size_t)k * HDIM + h] : wk[(size_t)k * HDIM + (h - HDIM)];
    o[idx] = (_Float16)v;
}

// dec_w2: [HDIM x GDIM] -> B^T [GDIM x HDIM]
__global__ void transpose_w_hxg(const float* __restrict__ w, _Float16* __restrict__ o) {
    int idx = blockIdx.x * blockDim.x + threadIdx.x;
    if (idx >= GDIM * HDIM) return;
    int n = idx / HDIM, k = idx % HDIM;
    o[idx] = (_Float16)w[(size_t)k * GDIM + n];
}

// ---------------------------------------------------------------- WMMA
// A fragment (16x32 f16): lane m(0-15)=row; lanes<16 hold K 0-15,
// lanes>=16 hold K 16-31.  B fragment (32x16): lane&15 = column,
// 16 contiguous K per lane (lanes<16: K 0-15, lanes>=16: K 16-31).
// C/D (16x16 f32): elem r -> M = r + 8*(lane>=16), N = lane&15.
__device__ __forceinline__ v16h load_a_frag(const _Float16* __restrict__ ap) {
    v8h alo = *(const v8h*)(ap);
    v8h ahi = *(const v8h*)(ap + 8);
    v16h a;
#pragma unroll
    for (int e = 0; e < 8; ++e) { a[e] = alo[e]; a[e + 8] = ahi[e]; }
    return a;
}

__device__ __forceinline__ v16h load_b_frag(const _Float16* __restrict__ bp) {
    v8h blo = *(const v8h*)(bp);
    v8h bhi = *(const v8h*)(bp + 8);
    v16h b;
#pragma unroll
    for (int e = 0; e < 8; ++e) { b[e] = blo[e]; b[e + 8] = bhi[e]; }
    return b;
}

__device__ __forceinline__ v8f wmma_acc(v16h a, v16h b, v8f acc) {
    return __builtin_amdgcn_wmma_f32_16x16x32_f16(false, a, false, b,
                                                  (short)0, acc, false, false);
}

// Fused enc1+kap1: C[M x HCMB] = A[M x KPAD] * BTcomb^T + bias.
// One wave owns a 16x64 output block (4 WMMA N-tiles share each A fragment).
// Software double-buffered: fragments for k+32 are in flight while the four
// WMMAs for k execute. blockIdx.z selects the wt / ko pass.
__global__ void __launch_bounds__(32)
gemm_enc_fused(const _Float16* __restrict__ Awt, const _Float16* __restrict__ Ako,
               const _Float16* __restrict__ BT,
               const float* __restrict__ enc_b1, const float* __restrict__ kap_b1,
               float* __restrict__ Cwt, float* __restrict__ Cko) {
    const _Float16* A = blockIdx.z ? Ako : Awt;
    float*          C = blockIdx.z ? Cko : Cwt;
    int lane = threadIdx.x;
    int m_base = blockIdx.x * 16;
    int n0 = blockIdx.y * 64;
    const _Float16* arow = A + (size_t)(m_base + (lane & 15)) * KPAD + ((lane < 16) ? 0 : 16);
    const _Float16* brow = BT + (size_t)(n0 + (lane & 15)) * KPAD + ((lane < 16) ? 0 : 16);

    v8f acc0 = {}, acc1 = {}, acc2 = {}, acc3 = {};
    v16h aC  = load_a_frag(arow);
    v16h b0C = load_b_frag(brow);
    v16h b1C = load_b_frag(brow + (size_t)16 * KPAD);
    v16h b2C = load_b_frag(brow + (size_t)32 * KPAD);
    v16h b3C = load_b_frag(brow + (size_t)48 * KPAD);

    for (int k0 = 0; k0 < KPAD; k0 += 32) {
        int kn = (k0 + 32 < KPAD) ? (k0 + 32) : k0;   // clamp: no OOB, no branch
        __builtin_prefetch(arow + k0 + 1024, 0, 1);
        v16h aN  = load_a_frag(arow + kn);
        v16h b0N = load_b_frag(brow + kn);
        v16h b1N = load_b_frag(brow + (size_t)16 * KPAD + kn);
        v16h b2N = load_b_frag(brow + (size_t)32 * KPAD + kn);
        v16h b3N = load_b_frag(brow + (size_t)48 * KPAD + kn);
        acc0 = wmma_acc(aC, b0C, acc0);
        acc1 = wmma_acc(aC, b1C, acc1);
        acc2 = wmma_acc(aC, b2C, acc2);
        acc3 = wmma_acc(aC, b3C, acc3);
        aC = aN; b0C = b0N; b1C = b1N; b2C = b2N; b3C = b3N;
    }

    int nlane = lane & 15;
    int mrow = m_base + ((lane < 16) ? 0 : 8);
    v8f accs[4] = {acc0, acc1, acc2, acc3};
#pragma unroll
    for (int t = 0; t < 4; ++t) {
        int n = n0 + t * 16 + nlane;
        float bv = (n < HDIM) ? enc_b1[n] : kap_b1[n - HDIM];
#pragma unroll
        for (int r = 0; r < 8; ++r)
            C[(size_t)(mrow + r) * HCMB + n] = accs[t][r] + bv;
    }
}

// Decoder GEMM fused with MSE: sum((A*B + bias - X)^2) -> atomic scalar.
// One wave owns 16x80 (5 N-tiles, 2000 = 25 groups * 80), double-buffered.
__global__ void __launch_bounds__(32)
gemm_dec2_mse(const _Float16* __restrict__ Awt, const _Float16* __restrict__ Ako,
              const _Float16* __restrict__ BT, const float* __restrict__ bias,
              const float* __restrict__ Xwt, const float* __restrict__ Xko,
              float* __restrict__ sq_sum) {
    const _Float16* A = blockIdx.z ? Ako : Awt;
    const float*    X = blockIdx.z ? Xko : Xwt;
    int lane = threadIdx.x;
    int m_base = blockIdx.x * 16;
    int n0 = blockIdx.y * 80;
    const _Float16* arow = A + (size_t)(m_base + (lane & 15)) * HDIM + ((lane < 16) ? 0 : 16);
    const _Float16* brow = BT + (size_t)(n0 + (lane & 15)) * HDIM + ((lane < 16) ? 0 : 16);

    v8f acc0 = {}, acc1 = {}, acc2 = {}, acc3 = {}, acc4 = {};
    v16h aC  = load_a_frag(arow);
    v16h b0C = load_b_frag(brow);
    v16h b1C = load_b_frag(brow + (size_t)16 * HDIM);
    v16h b2C = load_b_frag(brow + (size_t)32 * HDIM);
    v16h b3C = load_b_frag(brow + (size_t)48 * HDIM);
    v16h b4C = load_b_frag(brow + (size_t)64 * HDIM);

#pragma unroll
    for (int k0 = 0; k0 < HDIM; k0 += 32) {
        int kn = (k0 + 32 < HDIM) ? (k0 + 32) : k0;
        v16h aN  = load_a_frag(arow + kn);
        v16h b0N = load_b_frag(brow + kn);
        v16h b1N = load_b_frag(brow + (size_t)16 * HDIM + kn);
        v16h b2N = load_b_frag(brow + (size_t)32 * HDIM + kn);
        v16h b3N = load_b_frag(brow + (size_t)48 * HDIM + kn);
        v16h b4N = load_b_frag(brow + (size_t)64 * HDIM + kn);
        acc0 = wmma_acc(aC, b0C, acc0);
        acc1 = wmma_acc(aC, b1C, acc1);
        acc2 = wmma_acc(aC, b2C, acc2);
        acc3 = wmma_acc(aC, b3C, acc3);
        acc4 = wmma_acc(aC, b4C, acc4);
        aC = aN; b0C = b0N; b1C = b1N; b2C = b2N; b3C = b3N; b4C = b4N;
    }

    int nlane = lane & 15;
    int mrow = m_base + ((lane < 16) ? 0 : 8);
    float loc = 0.f;
    v8f accs[5] = {acc0, acc1, acc2, acc3, acc4};
#pragma unroll
    for (int t = 0; t < 5; ++t) {
        int n = n0 + t * 16 + nlane;
        float bv = bias[n];
#pragma unroll
        for (int r = 0; r < 8; ++r) {
            float d = accs[t][r] + bv - X[(size_t)(mrow + r) * GDIM + n];
            loc += d * d;
        }
    }
    loc = waveReduceSum(loc);
    if (lane == 0) atomicAdd(sq_sum, loc);
}

// ---------------------------------------------------------------- LayerNorm+ReLU
__global__ void __launch_bounds__(128)
ln_relu_f16(const float* __restrict__ H, int ld, int colofs,
            const float* __restrict__ g, const float* __restrict__ be,
            _Float16* __restrict__ O) {
    int row = blockIdx.x, t = threadIdx.x;
    float v = H[(size_t)row * ld + colofs + t];
    float mean = blockReduceSum128(v) * (1.f / HDIM);
    float d = v - mean;
    float var = blockReduceSum128(d * d) * (1.f / HDIM);
    float y = d * rsqrtf(var + 1e-5f) * g[t] + be[t];
    O[(size_t)row * HDIM + t] = (_Float16)fmaxf(y, 0.f);
}

// ---------------------------------------------------------------- small layers
__global__ void small_gemm_mu(const _Float16* __restrict__ Hf,
                              const float* __restrict__ w2,
                              const float* __restrict__ b2,
                              float* __restrict__ mu) {
    int idx = blockIdx.x * blockDim.x + threadIdx.x;
    if (idx >= NROWS * LDIM) return;
    int i = idx / LDIM, l = idx % LDIM;
    float s = b2[l];
    const _Float16* hr = Hf + (size_t)i * HDIM;
    for (int h = 0; h < HDIM; ++h) s += (float)hr[h] * w2[h * LDIM + l];
    mu[idx] = s;
}

__global__ void normalize_rows(const float* __restrict__ mu, float* __restrict__ z) {
    int i = blockIdx.x * blockDim.x + threadIdx.x;
    if (i >= NROWS) return;
    float v[LDIM], n2 = 0.f;
#pragma unroll
    for (int l = 0; l < LDIM; ++l) { v[l] = mu[i * LDIM + l]; n2 += v[l] * v[l]; }
    float inv = 1.f / fmaxf(sqrtf(n2), 1e-12f);
#pragma unroll
    for (int l = 0; l < LDIM; ++l) z[i * LDIM + l] = v[l] * inv;
}

__global__ void kappa_kernel(const _Float16* __restrict__ Hf,
                             const float* __restrict__ w2,
                             const float* __restrict__ b2,
                             float* __restrict__ kap_sum) {
    int i = blockIdx.x * blockDim.x + threadIdx.x;
    if (i >= NROWS) return;
    float s = b2[0];
    const _Float16* hr = Hf + (size_t)i * HDIM;
    for (int h = 0; h < HDIM; ++h) s += (float)hr[h] * w2[h];
    float sp = (s > 20.f) ? s : log1pf(__expf(s));
    atomicAdd(kap_sum, sp + 1e-6f);
}

__global__ void small_gemm_dec1(const float* __restrict__ z,
                                const float* __restrict__ w1,
                                const float* __restrict__ b1,
                                float* __restrict__ Hout) {
    int idx = blockIdx.x * blockDim.x + threadIdx.x;
    if (idx >= NROWS * HDIM) return;
    int i = idx / HDIM, h = idx % HDIM;
    float s = b1[h];
#pragma unroll
    for (int l = 0; l < LDIM; ++l) s += z[i * LDIM + l] * w1[l * HDIM + h];
    Hout[idx] = s;
}

// ---------------------------------------------------------------- contrastive
__global__ void __launch_bounds__(32)
contrast_lse(const float* __restrict__ zFix, const float* __restrict__ zIt,
             const int* __restrict__ yIt,
             float* __restrict__ lse_out, float* __restrict__ sum_out) {
    int i = blockIdx.x, lane = threadIdx.x;
    float zi[LDIM];
#pragma unroll
    for (int l = 0; l < LDIM; ++l) zi[l] = zFix[i * LDIM + l];
    float m[KCLS], s[KCLS], rs[KCLS];
#pragma unroll
    for (int c = 0; c < KCLS; ++c) { m[c] = -1e4f; s[c] = 0.f; rs[c] = 0.f; }
    for (int j = lane; j < NROWS; j += 32) {
        const float* zj = zIt + (size_t)j * LDIM;
        float d = 0.f;
#pragma unroll
        for (int l = 0; l < LDIM; ++l) d += zi[l] * zj[l];
        d *= INV_TAU;
        int cls = yIt[j];
#pragma unroll
        for (int c = 0; c < KCLS; ++c) {
            if (c == cls) {
                float M = fmaxf(m[c], d);
                s[c] = s[c] * __expf(m[c] - M) + __expf(d - M);
                m[c] = M;
                rs[c] += d;
            }
        }
    }
#pragma unroll
    for (int c = 0; c < KCLS; ++c) {
#pragma unroll
        for (int off = 16; off > 0; off >>= 1) {
            float m2 = __shfl_down(m[c], off, 32);
            float s2 = __shfl_down(s[c], off, 32);
            float r2 = __shfl_down(rs[c], off, 32);
            float M = fmaxf(m[c], m2);
            s[c] = s[c] * __expf(m[c] - M) + s2 * __expf(m2 - M);
            m[c] = M;
            rs[c] += r2;
        }
        if (lane == 0) {
            lse_out[i * KCLS + c] = m[c] + __logf(fmaxf(s[c], 1e-30f));
            sum_out[i * KCLS + c] = rs[c];
        }
    }
}

__global__ void count_labels(const int* __restrict__ yw, const int* __restrict__ yk,
                             int* cw, int* ck, int* ca) {
    int i = blockIdx.x * blockDim.x + threadIdx.x;
    if (i < NROWS) { atomicAdd(&cw[yw[i]], 1); atomicAdd(&ca[yw[i]], 1); }
    else if (i < 2 * NROWS) {
        int j = i - NROWS;
        atomicAdd(&ck[yk[j]], 1); atomicAdd(&ca[yk[j]], 1);
    }
}

__global__ void __launch_bounds__(256)
contrast_reduce(const int* __restrict__ yw, const int* __restrict__ yk,
                const int* __restrict__ cw, const int* __restrict__ ck,
                const float* __restrict__ lse_r, const float* __restrict__ rsum,
                const float* __restrict__ lse_c, const float* __restrict__ csum,
                float* ctotal, float* nshared) {
    int c = blockIdx.x, t = threadIdx.x;
    float nw = fmaxf((float)cw[c], 1.f), nk = fmaxf((float)ck[c], 1.f);
    float p1 = 0.f, p2 = 0.f;
    for (int i = t; i < NROWS; i += 256) {
        if (yw[i] == c) p1 += lse_r[i * KCLS + c] - rsum[i * KCLS + c] / nk;
        if (yk[i] == c) p2 += lse_c[i * KCLS + c] - csum[i * KCLS + c] / nw;
    }
    p1 = waveReduceSum(p1);
    p2 = waveReduceSum(p2);
    __shared__ float sm1[8], sm2[8];
    int lane = t & 31, w = t >> 5;
    if (lane == 0) { sm1[w] = p1; sm2[w] = p2; }
    __syncthreads();
    if (t == 0) {
        float l1 = 0.f, l2 = 0.f;
        for (int k = 0; k < 8; ++k) { l1 += sm1[k]; l2 += sm2[k]; }
        bool valid = (cw[c] > 0) && (ck[c] > 0);
        float term = valid ? 0.5f * (l1 / nw + l2 / nk) : 0.f;
        atomicAdd(ctotal, term);
        atomicAdd(nshared, valid ? 1.f : 0.f);
    }
}

// ---------------------------------------------------------------- alignment
__global__ void align_sums(const float* __restrict__ zwt, const float* __restrict__ zko,
                           const int* __restrict__ yw, const int* __restrict__ yk,
                           float* __restrict__ sums) {
    int idx = blockIdx.x * blockDim.x + threadIdx.x;
    if (idx >= 2 * NROWS * LDIM) return;
    int sl = idx / LDIM, l = idx % LDIM;
    int c; float zv;
    if (sl < NROWS) { c = yw[sl]; zv = zwt[sl * LDIM + l]; }
    else { int j = sl - NROWS; c = yk[j]; zv = zko[j * LDIM + l]; }
    atomicAdd(&sums[c * LDIM + l], zv);
}

__global__ void align_dev(const float* __restrict__ zwt, const float* __restrict__ zko,
                          const int* __restrict__ yw, const int* __restrict__ yk,
                          const float* __restrict__ sums, const int* __restrict__ ca,
                          float* __restrict__ devc) {
    int sl = blockIdx.x * blockDim.x + threadIdx.x;
    if (sl >= 2 * NROWS) return;
    const float* z; int c;
    if (sl < NROWS) { z = zwt + (size_t)sl * LDIM; c = yw[sl]; }
    else { z = zko + (size_t)(sl - NROWS) * LDIM; c = yk[sl - NROWS]; }
    float cnt = fmaxf((float)ca[c], 1.f);
    float acc = 0.f;
#pragma unroll
    for (int l = 0; l < LDIM; ++l) {
        float d = z[l] - sums[c * LDIM + l] / cnt;
        acc += d * d;
    }
    atomicAdd(&devc[c], acc);
}

// ---------------------------------------------------------------- final
__global__ void final_combine(const float* __restrict__ scal,
                              const int* __restrict__ ca,
                              const float* __restrict__ devc,
                              float* __restrict__ out) {
    if (threadIdx.x == 0 && blockIdx.x == 0) {
        float recon = scal[0] / ((float)NROWS * (float)GDIM);
        float kl = 0.01f * scal[1] / (float)NROWS;
        float contrast = (scal[3] > 0.f) ? scal[2] / fmaxf(scal[3], 1.f) : 0.f;
        float align = 0.f, np = 0.f;
        for (int c = 0; c < KCLS; ++c)
            if (ca[c] > 0) { align += devc[c] / fmaxf((float)ca[c], 1.f); np += 1.f; }
        align = align / fmaxf(np, 1.f);
        out[0] = recon + kl + 1.0f * contrast + 0.5f * align;
    }
}

// ---------------------------------------------------------------- launch
extern "C" void kernel_launch(void* const* d_in, const int* in_sizes, int n_in,
                              void* d_out, int out_size, void* d_ws, size_t ws_size,
                              hipStream_t stream) {
    (void)in_sizes; (void)n_in; (void)out_size; (void)ws_size;
    const float* wt_x   = (const float*)d_in[0];
    const float* ko_x   = (const float*)d_in[1];
    const int*   ywt    = (const int*)d_in[2];
    const int*   yko    = (const int*)d_in[3];
    const float* enc_w1 = (const float*)d_in[4];
    const float* enc_b1 = (const float*)d_in[5];
    const float* enc_g  = (const float*)d_in[6];
    const float* enc_be = (const float*)d_in[7];
    const float* enc_w2 = (const float*)d_in[8];
    const float* enc_b2 = (const float*)d_in[9];
    const float* kap_w1 = (const float*)d_in[10];
    const float* kap_b1 = (const float*)d_in[11];
    const float* kap_g  = (const float*)d_in[12];
    const float* kap_be = (const float*)d_in[13];
    const float* kap_w2 = (const float*)d_in[14];
    const float* kap_b2 = (const float*)d_in[15];
    const float* dec_w1 = (const float*)d_in[16];
    const float* dec_b1 = (const float*)d_in[17];
    const float* dec_g  = (const float*)d_in[18];
    const float* dec_be = (const float*)d_in[19];
    const float* dec_w2 = (const float*)d_in[20];
    const float* dec_b2 = (const float*)d_in[21];

    char* base = (char*)d_ws;
    size_t off = 0;
    auto alloc = [&](size_t bytes) -> void* {
        void* r = base + off;
        off = (off + bytes + 255) & ~(size_t)255;
        return r;
    };
    _Float16* wtX16    = (_Float16*)alloc((size_t)NROWS * KPAD * 2);
    _Float16* koX16    = (_Float16*)alloc((size_t)NROWS * KPAD * 2);
    _Float16* BTcomb   = (_Float16*)alloc((size_t)HCMB * KPAD * 2);
    _Float16* decW2T   = (_Float16*)alloc((size_t)GDIM * HDIM * 2);
    float*    wtH      = (float*)alloc((size_t)NROWS * HCMB * 4);
    float*    koH      = (float*)alloc((size_t)NROWS * HCMB * 4);
    _Float16* wtEncH16 = (_Float16*)alloc((size_t)NROWS * HDIM * 2);
    _Float16* koEncH16 = (_Float16*)alloc((size_t)NROWS * HDIM * 2);
    _Float16* wtKapH16 = (_Float16*)alloc((size_t)NROWS * HDIM * 2);
    _Float16* koKapH16 = (_Float16*)alloc((size_t)NROWS * HDIM * 2);
    float*    wtMu     = (float*)alloc((size_t)NROWS * LDIM * 4);
    float*    koMu     = (float*)alloc((size_t)NROWS * LDIM * 4);
    float*    zwt      = (float*)alloc((size_t)NROWS * LDIM * 4);
    float*    zko      = (float*)alloc((size_t)NROWS * LDIM * 4);
    float*    wtDecH   = (float*)alloc((size_t)NROWS * HDIM * 4);
    float*    koDecH   = (float*)alloc((size_t)NROWS * HDIM * 4);
    _Float16* wtDecH16 = (_Float16*)alloc((size_t)NROWS * HDIM * 2);
    _Float16* koDecH16 = (_Float16*)alloc((size_t)NROWS * HDIM * 2);
    float*    lse_r    = (float*)alloc((size_t)NROWS * KCLS * 4);
    float*    rsum     = (float*)alloc((size_t)NROWS * KCLS * 4);
    float*    lse_c    = (float*)alloc((size_t)NROWS * KCLS * 4);
    float*    csum     = (float*)alloc((size_t)NROWS * KCLS * 4);
    float*    scal     = (float*)alloc(64);
    int*      cw       = (int*)alloc(64);
    int*      ck       = (int*)alloc(64);
    int*      ca       = (int*)alloc(64);
    float*    sums     = (float*)alloc(KCLS * LDIM * 4);
    float*    devc     = (float*)alloc(64);

    init_accum<<<1, 128, 0, stream>>>(scal, cw, ck, ca, sums, devc);

    int T = 256;
    cast_pad_x<<<(NROWS * KPAD + T - 1) / T, T, 0, stream>>>(wt_x, wtX16);
    cast_pad_x<<<(NROWS * KPAD + T - 1) / T, T, 0, stream>>>(ko_x, koX16);
    transpose_w_pair<<<(HCMB * KPAD + T - 1) / T, T, 0, stream>>>(enc_w1, kap_w1, BTcomb);
    transpose_w_hxg<<<(GDIM * HDIM + T - 1) / T, T, 0, stream>>>(dec_w2, decW2T);

    dim3 g1(NROWS / 16, HCMB / 64, 2);
    gemm_enc_fused<<<g1, 32, 0, stream>>>(wtX16, koX16, BTcomb,
                                          enc_b1, kap_b1, wtH, koH);

    ln_relu_f16<<<NROWS, 128, 0, stream>>>(wtH, HCMB, 0,    enc_g, enc_be, wtEncH16);
    ln_relu_f16<<<NROWS, 128, 0, stream>>>(koH, HCMB, 0,    enc_g, enc_be, koEncH16);
    ln_relu_f16<<<NROWS, 128, 0, stream>>>(wtH, HCMB, HDIM, kap_g, kap_be, wtKapH16);
    ln_relu_f16<<<NROWS, 128, 0, stream>>>(koH, HCMB, HDIM, kap_g, kap_be, koKapH16);

    small_gemm_mu<<<(NROWS * LDIM + T - 1) / T, T, 0, stream>>>(wtEncH16, enc_w2, enc_b2, wtMu);
    small_gemm_mu<<<(NROWS * LDIM + T - 1) / T, T, 0, stream>>>(koEncH16, enc_w2, enc_b2, koMu);
    normalize_rows<<<(NROWS + T - 1) / T, T, 0, stream>>>(wtMu, zwt);
    normalize_rows<<<(NROWS + T - 1) / T, T, 0, stream>>>(koMu, zko);

    kappa_kernel<<<(NROWS + T - 1) / T, T, 0, stream>>>(wtKapH16, kap_w2, kap_b2, scal + 1);
    kappa_kernel<<<(NROWS + T - 1) / T, T, 0, stream>>>(koKapH16, kap_w2, kap_b2, scal + 1);

    small_gemm_dec1<<<(NROWS * HDIM + T - 1) / T, T, 0, stream>>>(zwt, dec_w1, dec_b1, wtDecH);
    small_gemm_dec1<<<(NROWS * HDIM + T - 1) / T, T, 0, stream>>>(zko, dec_w1, dec_b1, koDecH);
    ln_relu_f16<<<NROWS, 128, 0, stream>>>(wtDecH, HDIM, 0, dec_g, dec_be, wtDecH16);
    ln_relu_f16<<<NROWS, 128, 0, stream>>>(koDecH, HDIM, 0, dec_g, dec_be, koDecH16);

    dim3 g2(NROWS / 16, GDIM / 80, 2);   // 25 groups of 5 N-tiles
    gemm_dec2_mse<<<g2, 32, 0, stream>>>(wtDecH16, koDecH16, decW2T, dec_b2,
                                         wt_x, ko_x, scal + 0);

    count_labels<<<(2 * NROWS + T - 1) / T, T, 0, stream>>>(ywt, yko, cw, ck, ca);
    contrast_lse<<<NROWS, 32, 0, stream>>>(zwt, zko, yko, lse_r, rsum);
    contrast_lse<<<NROWS, 32, 0, stream>>>(zko, zwt, ywt, lse_c, csum);
    contrast_reduce<<<KCLS, 256, 0, stream>>>(ywt, yko, cw, ck, lse_r, rsum,
                                              lse_c, csum, scal + 2, scal + 3);

    align_sums<<<(2 * NROWS * LDIM + T - 1) / T, T, 0, stream>>>(zwt, zko, ywt, yko, sums);
    align_dev<<<(2 * NROWS + T - 1) / T, T, 0, stream>>>(zwt, zko, ywt, yko, sums, ca, devc);

    final_combine<<<1, 1, 0, stream>>>(scal, ca, devc, (float*)d_out);
}
